// CustomDynaRNN_47906065219674
// MI455X (gfx1250) — matrix operand order repaired and checked by reading, not compiled, not run on hardware
//
#include <hip/hip_runtime.h>

typedef __bf16 bf16_t;
typedef bf16_t v16bf __attribute__((ext_vector_type(16)));
typedef bf16_t v8bf  __attribute__((ext_vector_type(8)));
typedef float  v8f   __attribute__((ext_vector_type(8)));

// Problem constants
constexpr int kB   = 256;
constexpr int kT   = 512;
constexpr int kEMB = 64;
constexpr int kOB  = 64;
constexpr int kAC  = 10;
constexpr int kH   = 128;
constexpr int kSA  = 288;   // A staging row stride (bf16 elems), >= max padded K

// K-chunk (32-wide) counts per matmul, and N tile (16-wide) counts
constexpr int kKT1  = 3;   // Wt1: K=74  -> 96
constexpr int kKT2  = 8;   // Wt2: K=256
constexpr int kKIH0 = 5;   // W_ih0: K=138 -> 160
constexpr int kKHH  = 4;   // W_hh*, W_ih1: K=128
constexpr int kKD1  = 9;   // Wd1: K=266 -> 288
constexpr int kKMU  = 8;   // Wmu/Wstd: K=256
constexpr int kNT1  = 16;  // N=256
constexpr int kNTG  = 24;  // N=384
constexpr int kNTD1 = 16;  // N=256
constexpr int kNTMU = 4;   // N=64
constexpr int kNT2  = 4;   // N=64

// Opaque pointer: blocks LICM/register-caching of loop-invariant weight loads.
__device__ __forceinline__ const bf16_t* opaque(const bf16_t* p) {
    asm volatile("" : "+s"(p));
    return p;
}

// Explicit global-address-space 32B load (restores global_load_b128 after the
// opaque() barrier hides the pointer's provenance from addrspace inference).
__device__ __forceinline__ v16bf load_bfrag_global(const bf16_t* p) {
    const __attribute__((address_space(1))) v16bf* gp =
        (const __attribute__((address_space(1))) v16bf*)(unsigned long long)p;
    return *gp;
}

// ---------------------------------------------------------------------------
// Branch-free fast transcendentals on native v_exp_f32 / v_log_f32 / v_rcp_f32
// ---------------------------------------------------------------------------
__device__ __forceinline__ float fast_exp(float x) {
    return __builtin_amdgcn_exp2f(x * 1.4426950408889634f);
}
__device__ __forceinline__ float fast_sigmoid(float x) {
    return __builtin_amdgcn_rcpf(1.f + fast_exp(-x));
}
__device__ __forceinline__ float fast_tanh(float x) {
    return 2.f * __builtin_amdgcn_rcpf(1.f + fast_exp(-2.f * x)) - 1.f;
}
__device__ __forceinline__ float fast_softplus(float x) {
    float e = __builtin_amdgcn_exp2f(-fabsf(x) * 1.4426950408889634f);
    return fmaxf(x, 0.f) + 0.6931471805599453f * __builtin_amdgcn_logf(1.f + e);
}

// ---------------------------------------------------------------------------
// Weight repack: fragment-major bf16 B-operands for v_wmma_f32_16x16x32_bf16.
// Fragment (ntile, kch) = 512 bf16 = [lane][s]:
//   n = ntile*16 + lane%16 ; k = kch*32 + 8*(lane/16) + (s%8) + 16*(s/8)   (s = 0..15)
// ---------------------------------------------------------------------------
__global__ void prep_frags(const float* __restrict__ src, bf16_t* __restrict__ dst,
                           int K, int N, int KCH, int NT, int transposed)
{
    int total = KCH * NT * 512;
    for (int idx = blockIdx.x * blockDim.x + threadIdx.x; idx < total;
         idx += gridDim.x * blockDim.x) {
        int frag   = idx >> 9;
        int within = idx & 511;
        int lane   = within >> 4;
        int s      = within & 15;
        int ntile  = frag / KCH;
        int kch    = frag - ntile * KCH;
        int n      = ntile * 16 + (lane & 15);
        int k      = kch * 32 + ((lane >> 4) << 3) + (s & 7) + ((s >> 3) << 4);
        float v = 0.f;
        if (k < K && n < N)
            v = transposed ? src[(size_t)n * K + k] : src[(size_t)k * N + n];
        dst[idx] = (bf16_t)v;
    }
}

// ---------------------------------------------------------------------------
// One 16x16 output tile: accumulate over kch 32-deep WMMA steps.
// ---------------------------------------------------------------------------
__device__ __forceinline__ v8f mm_rowtile(const bf16_t* __restrict__ bfrag,
                                          const bf16_t* sAbase, int astride,
                                          int kch_count, int lane)
{
    v8f acc = {};
    const int row  = lane & 15;
    const int half = lane >> 4;
    const bf16_t* ap = sAbase + row * astride + half * 8;
#pragma unroll
    for (int kc = 0; kc < kch_count; ++kc) {
        v8bf lo = *(const v8bf*)(ap + kc * 32);
        v8bf hi = *(const v8bf*)(ap + kc * 32 + 16);
        v16bf a = __builtin_shufflevector(lo, hi, 0,1,2,3,4,5,6,7,8,9,10,11,12,13,14,15);
        v16bf b = load_bfrag_global(bfrag + (size_t)kc * 512 + lane * 16);
        acc = __builtin_amdgcn_wmma_f32_16x16x32_bf16(false, a, false, b,
                                                      (short)0, acc, false, false);
    }
    return acc;
}

// D fragment -> f32 LDS buffer (row-major, width nfull), + bias
__device__ __forceinline__ void store_tile_f32(float* dst, int nfull, int ntile,
                                               int lane, v8f acc, float bias)
{
    int n  = ntile * 16 + (lane & 15);
    int mb = (lane >> 4) * 8;
#pragma unroll
    for (int v = 0; v < 8; ++v)
        dst[(mb + v) * nfull + n] = acc[v] + bias;
}

// D fragment -> bf16 LDS A-staging buffer with tanh
__device__ __forceinline__ void store_tile_bf_tanh(bf16_t* dst, int stride, int ntile,
                                                   int lane, v8f acc, float bias)
{
    int n  = ntile * 16 + (lane & 15);
    int mb = (lane >> 4) * 8;
#pragma unroll
    for (int v = 0; v < 8; ++v)
        dst[(mb + v) * stride + n] = (bf16_t)fast_tanh(acc[v] + bias);
}

// ---------------------------------------------------------------------------
// Main persistent kernel: one workgroup = 16 batch rows, full T loop.
// ---------------------------------------------------------------------------
__global__ __launch_bounds__(256, 1)
void dyna_rnn_kernel(const float* __restrict__ ob_emb, const float* __restrict__ ac,
                     const float* __restrict__ noise, const float* __restrict__ prevh,
                     const int* __restrict__ seqlen,
                     const float* __restrict__ b_ih0, const float* __restrict__ b_hh0,
                     const float* __restrict__ b_ih1, const float* __restrict__ b_hh1,
                     const float* __restrict__ bt1, const float* __restrict__ bt2,
                     const float* __restrict__ bd1, const float* __restrict__ bmu,
                     const float* __restrict__ bstd,
                     const bf16_t* __restrict__ fWt1, const bf16_t* __restrict__ fWt2,
                     const bf16_t* __restrict__ fWih0, const bf16_t* __restrict__ fWhh0,
                     const bf16_t* __restrict__ fWih1, const bf16_t* __restrict__ fWhh1,
                     const bf16_t* __restrict__ fWd1, const bf16_t* __restrict__ fWmu,
                     const bf16_t* __restrict__ fWstd,
                     float* __restrict__ out_mu, float* __restrict__ out_std,
                     float* __restrict__ out_ret, float* __restrict__ out_fin)
{
    __shared__ __align__(16) bf16_t sA [16 * kSA];   // A staging (K up to 288)
    __shared__ __align__(16) bf16_t sA2[16 * 256];   // second A buffer (t1/h/nh0/hdec)
    __shared__ float sH0 [16 * kH];
    __shared__ float sH1 [16 * kH];
    __shared__ float sOs [16 * kOB];                 // ob_sim
    __shared__ float sNos[16 * kOB];                 // next_ob_sim
    __shared__ float sAc [16 * kAC];
    __shared__ float sObT[16 * kEMB];
    __shared__ float sEps[16 * kOB];
    __shared__ float sMu [16 * kOB];
    __shared__ float sSd [16 * kOB];

    const int tid   = threadIdx.x;
    const int lane  = tid & 31;
    const int wave  = tid >> 5;
    const int bbase = blockIdx.x * 16;

    // ---- init recurrent state from prev_hidden_states ----
    for (int i = tid; i < 16 * 320; i += 256) {
        int r = i / 320, c = i % 320;
        float v = prevh[(size_t)(bbase + r) * 320 + c];
        if (c < kH)           sH0[r * kH + c] = v;
        else if (c < 2 * kH)  sH1[r * kH + (c - kH)] = v;
        else                  sOs[r * kOB + (c - 2 * kH)] = v;
    }

    // ---- hoist loop-invariant scalars into registers ----
    const int jc  = wave * 16 + (lane & 15);          // gate column for this lane
    const int nc1 = (wave + 8) * 16 + (lane & 15);    // second 256-wide tile column
    const float c_bi0r = b_ih0[jc],      c_bi0z = b_ih0[kH + jc], c_bi0n = b_ih0[2*kH + jc];
    const float c_bh0r = b_hh0[jc],      c_bh0z = b_hh0[kH + jc], c_bh0n = b_hh0[2*kH + jc];
    const float c_bi1r = b_ih1[jc],      c_bi1z = b_ih1[kH + jc], c_bi1n = b_ih1[2*kH + jc];
    const float c_bh1r = b_hh1[jc],      c_bh1z = b_hh1[kH + jc], c_bh1n = b_hh1[2*kH + jc];
    const float c_bt1a = bt1[jc],        c_bt1b = bt1[nc1];
    const float c_bd1a = bd1[jc],        c_bd1b = bd1[nc1];
    const float c_bt2  = (wave < 4) ? bt2[wave * 16 + (lane & 15)] : 0.f;
    const float c_bmu  = (wave < 4) ? bmu[wave * 16 + (lane & 15)] : 0.f;
    const float c_bsd  = (wave >= 4) ? bstd[(wave - 4) * 16 + (lane & 15)] : 0.f;
    int seq_m[8];
#pragma unroll
    for (int v = 0; v < 8; ++v)
        seq_m[v] = seqlen[bbase + (lane >> 4) * 8 + v];
    __syncthreads();

    for (int t = 0; t < kT; ++t) {
        // Fresh (non-hoistable) views of the shared weight-fragment arrays.
        const bf16_t* pWt1  = opaque(fWt1);
        const bf16_t* pWt2  = opaque(fWt2);
        const bf16_t* pWih0 = opaque(fWih0);
        const bf16_t* pWhh0 = opaque(fWhh0);
        const bf16_t* pWih1 = opaque(fWih1);
        const bf16_t* pWhh1 = opaque(fWhh1);
        const bf16_t* pWd1  = opaque(fWd1);
        const bf16_t* pWmu  = opaque(fWmu);
        const bf16_t* pWstd = opaque(fWstd);

        // -- Phase 1: per-step inputs (ac, ob_t, eps) + prefetch next step --
        for (int i = tid; i < 16 * kAC; i += 256) {
            int r = i / kAC, c = i % kAC;
            sAc[i] = ac[((size_t)(bbase + r) * kT + t) * kAC + c];
        }
        for (int i = tid; i < 16 * kEMB; i += 256) {
            int r = i >> 6, c = i & 63;
            size_t bt_ = (size_t)(bbase + r) * kT + t;
            sObT[i] = ob_emb[bt_ * kEMB + c];
            sEps[i] = noise [bt_ * kOB  + c];
            if (t + 1 < kT) {
                __builtin_prefetch(&ob_emb[(bt_ + 1) * kEMB + c], 0, 1);
                __builtin_prefetch(&noise [(bt_ + 1) * kOB  + c], 0, 1);
            }
        }
        // -- Phase 2: stage A = [ob_sim(64) | ac(10) | 0] K=96 --
        for (int i = tid; i < 16 * 96; i += 256) {
            int r = i / 96, k = i % 96;
            float v = (k < kOB) ? sOs[r * kOB + k]
                    : (k < kOB + kAC ? sAc[r * kAC + (k - kOB)] : 0.f);
            sA[r * kSA + k] = (bf16_t)v;
        }
        __syncthreads();

        // -- Phase 3: t1 = tanh(A @ Wt1 + bt1)  -> sA2 (bf16, 16x256) --
        {
            v8f a0 = mm_rowtile(pWt1 + (size_t)wave       * kKT1 * 512, sA, kSA, kKT1, lane);
            v8f a1 = mm_rowtile(pWt1 + (size_t)(wave + 8) * kKT1 * 512, sA, kSA, kKT1, lane);
            store_tile_bf_tanh(sA2, 256, wave,     lane, a0, c_bt1a);
            store_tile_bf_tanh(sA2, 256, wave + 8, lane, a1, c_bt1b);
        }
        __syncthreads();

        // -- Phase 4: next_ob_sim = t1 @ Wt2 + bt2 -> sNos (waves 0-3) --
        if (wave < kNT2) {
            v8f a = mm_rowtile(pWt2 + (size_t)wave * kKT2 * 512, sA2, 256, kKT2, lane);
            store_tile_f32(sNos, kOB, wave, lane, a, c_bt2);
        }
        __syncthreads();

        // -- Phase 5: stage x=[ob_t|nos|ac|0] K=160 into sA ; h0 bf16 into sA2 --
        for (int i = tid; i < 16 * 160; i += 256) {
            int r = i / 160, k = i % 160;
            float v;
            if (k < kEMB)                  v = sObT[r * kEMB + k];
            else if (k < kEMB + kOB)       v = sNos[r * kOB + (k - kEMB)];
            else if (k < kEMB + kOB + kAC) v = sAc[r * kAC + (k - kEMB - kOB)];
            else                           v = 0.f;
            sA[r * kSA + k] = (bf16_t)v;
        }
        for (int i = tid; i < 16 * kH; i += 256) {
            int r = i >> 7, j = i & 127;
            sA2[r * 256 + j] = (bf16_t)sH0[i];
        }
        __syncthreads();

        // -- Phase 6/7: GRU layer 0, gate fragments register-resident --
        {
            v8f gir = mm_rowtile(pWih0 + (size_t)(wave     ) * kKIH0 * 512, sA,  kSA, kKIH0, lane);
            v8f giz = mm_rowtile(pWih0 + (size_t)(wave + 8 ) * kKIH0 * 512, sA,  kSA, kKIH0, lane);
            v8f gin = mm_rowtile(pWih0 + (size_t)(wave + 16) * kKIH0 * 512, sA,  kSA, kKIH0, lane);
            v8f ghr = mm_rowtile(pWhh0 + (size_t)(wave     ) * kKHH  * 512, sA2, 256, kKHH, lane);
            v8f ghz = mm_rowtile(pWhh0 + (size_t)(wave + 8 ) * kKHH  * 512, sA2, 256, kKHH, lane);
            v8f ghn = mm_rowtile(pWhh0 + (size_t)(wave + 16) * kKHH  * 512, sA2, 256, kKHH, lane);
            __syncthreads();
#pragma unroll
            for (int v = 0; v < 8; ++v) {
                int m = (lane >> 4) * 8 + v;
                float rg  = fast_sigmoid((gir[v] + c_bi0r) + (ghr[v] + c_bh0r));
                float zg  = fast_sigmoid((giz[v] + c_bi0z) + (ghz[v] + c_bh0z));
                float ng  = fast_tanh((gin[v] + c_bi0n) + rg * (ghn[v] + c_bh0n));
                float nh0 = (1.f - zg) * ng + zg * sH0[m * kH + jc];
                float msk = (t < seq_m[v]) ? 1.f : 0.f;
                sA2[m * 256 + jc] = (bf16_t)nh0;              // unmasked nh0 -> layer 1
                sH0[m * kH + jc]  = msk * nh0;
                sA[m * kSA + jc]  = (bf16_t)sH1[m * kH + jc]; // stage old h1
            }
            __syncthreads();
        }

        // -- Phase 8/9: GRU layer 1 (gi from nh0, gh from old h1) --
        {
            v8f gir = mm_rowtile(pWih1 + (size_t)(wave     ) * kKHH * 512, sA2, 256, kKHH, lane);
            v8f giz = mm_rowtile(pWih1 + (size_t)(wave + 8 ) * kKHH * 512, sA2, 256, kKHH, lane);
            v8f gin = mm_rowtile(pWih1 + (size_t)(wave + 16) * kKHH * 512, sA2, 256, kKHH, lane);
            v8f ghr = mm_rowtile(pWhh1 + (size_t)(wave     ) * kKHH * 512, sA,  kSA, kKHH, lane);
            v8f ghz = mm_rowtile(pWhh1 + (size_t)(wave + 8 ) * kKHH * 512, sA,  kSA, kKHH, lane);
            v8f ghn = mm_rowtile(pWhh1 + (size_t)(wave + 16) * kKHH * 512, sA,  kSA, kKHH, lane);
            __syncthreads();
#pragma unroll
            for (int v = 0; v < 8; ++v) {
                int m = (lane >> 4) * 8 + v;
                float rg  = fast_sigmoid((gir[v] + c_bi1r) + (ghr[v] + c_bh1r));
                float zg  = fast_sigmoid((giz[v] + c_bi1z) + (ghz[v] + c_bh1z));
                float ng  = fast_tanh((gin[v] + c_bi1n) + rg * (ghn[v] + c_bh1n));
                float nh1 = (1.f - zg) * ng + zg * sH1[m * kH + jc];
                float msk = (t < seq_m[v]) ? 1.f : 0.f;
                float ov  = msk * nh1;                        // out == new h1
                sH1[m * kH + jc] = ov;
                sA[m * kSA + jc] = (bf16_t)ov;                // hdec input cols 0..127
            }
            // hdec input cols 128..287 : [nos(64) | ob_t(64) | ac(10) | 0]
            for (int i = tid; i < 16 * 160; i += 256) {
                int r = i / 160, k = i % 160;
                float v;
                if (k < kOB)                   v = sNos[r * kOB + k];
                else if (k < kOB + kEMB)       v = sObT[r * kEMB + (k - kOB)];
                else if (k < kOB + kEMB + kAC) v = sAc[r * kAC + (k - kOB - kEMB)];
                else                           v = 0.f;
                sA[r * kSA + kH + k] = (bf16_t)v;
            }
            __syncthreads();
        }

        // -- Phase 10: hdec = tanh(A @ Wd1 + bd1) -> sA2 (bf16) --
        {
            v8f a0 = mm_rowtile(pWd1 + (size_t)wave       * kKD1 * 512, sA, kSA, kKD1, lane);
            v8f a1 = mm_rowtile(pWd1 + (size_t)(wave + 8) * kKD1 * 512, sA, kSA, kKD1, lane);
            store_tile_bf_tanh(sA2, 256, wave,     lane, a0, c_bd1a);
            store_tile_bf_tanh(sA2, 256, wave + 8, lane, a1, c_bd1b);
        }
        __syncthreads();

        // -- Phase 11: mu (waves 0-3) and std (waves 4-7) heads --
        if (wave < 4) {
            v8f a = mm_rowtile(pWmu + (size_t)wave * kKMU * 512, sA2, 256, kKMU, lane);
            store_tile_f32(sMu, kOB, wave, lane, a, c_bmu);
        } else {
            int nt = wave - 4;
            v8f a = mm_rowtile(pWstd + (size_t)nt * kKMU * 512, sA2, 256, kKMU, lane);
            int n  = nt * 16 + (lane & 15);
            int mb = (lane >> 4) * 8;
#pragma unroll
            for (int v = 0; v < 8; ++v)
                sSd[(mb + v) * kOB + n] = fast_softplus(a[v] + c_bsd) + 1e-4f;
        }
        __syncthreads();

        // -- Phase 12: ob_sim update + global stores --
        for (int i = tid; i < 16 * kOB; i += 256) {
            int r = i >> 6, c = i & 63;
            float mu = sMu[i], sd = sSd[i];
            float os = mu + sd * sEps[i];
            sOs[i] = os;
            size_t bt_ = (size_t)(bbase + r) * kT + t;
            out_mu [bt_ * kOB + c] = mu;
            out_std[bt_ * kOB + c] = sd;
            out_ret[bt_ * 320 + 256 + c] = os;
        }
        for (int i = tid; i < 16 * 256; i += 256) {
            int r = i >> 8, j = i & 255;
            size_t bt_ = (size_t)(bbase + r) * kT + t;
            out_ret[bt_ * 320 + j] = (j < kH) ? sH0[r * kH + j] : sH1[r * kH + (j - kH)];
        }
        __syncthreads();
    }

    // final state = rets[:, -1, :]
    for (int i = tid; i < 16 * 320; i += 256) {
        int r = i / 320, c = i % 320;
        float v = (c < kH) ? sH0[r * kH + c]
                : (c < 2 * kH ? sH1[r * kH + (c - kH)] : sOs[r * kOB + (c - 2 * kH)]);
        out_fin[(size_t)(bbase + r) * 320 + c] = v;
    }
}

// ---------------------------------------------------------------------------
extern "C" void kernel_launch(void* const* d_in, const int* in_sizes, int n_in,
                              void* d_out, int out_size, void* d_ws, size_t ws_size,
                              hipStream_t stream)
{
    const float* ob_emb = (const float*)d_in[0];
    const float* ac     = (const float*)d_in[1];
    const float* noise  = (const float*)d_in[2];
    const float* prevh  = (const float*)d_in[3];
    const int*   seq    = (const int*)  d_in[4];
    const float* W_ih0  = (const float*)d_in[5];
    const float* W_hh0  = (const float*)d_in[6];
    const float* b_ih0  = (const float*)d_in[7];
    const float* b_hh0  = (const float*)d_in[8];
    const float* W_ih1  = (const float*)d_in[9];
    const float* W_hh1  = (const float*)d_in[10];
    const float* b_ih1  = (const float*)d_in[11];
    const float* b_hh1  = (const float*)d_in[12];
    const float* Wt1    = (const float*)d_in[13];
    const float* bt1    = (const float*)d_in[14];
    const float* Wt2    = (const float*)d_in[15];
    const float* bt2    = (const float*)d_in[16];
    const float* Wd1    = (const float*)d_in[17];
    const float* bd1    = (const float*)d_in[18];
    const float* Wmu    = (const float*)d_in[19];
    const float* bmu    = (const float*)d_in[20];
    const float* Wstd   = (const float*)d_in[21];
    const float* bstd   = (const float*)d_in[22];

    bf16_t* ws = (bf16_t*)d_ws;
    size_t off = 0;
    bf16_t* fWt1  = ws + off; off += (size_t)kKT1  * kNT1  * 512;
    bf16_t* fWt2  = ws + off; off += (size_t)kKT2  * kNT2  * 512;
    bf16_t* fWih0 = ws + off; off += (size_t)kKIH0 * kNTG  * 512;
    bf16_t* fWhh0 = ws + off; off += (size_t)kKHH  * kNTG  * 512;
    bf16_t* fWih1 = ws + off; off += (size_t)kKHH  * kNTG  * 512;
    bf16_t* fWhh1 = ws + off; off += (size_t)kKHH  * kNTG  * 512;
    bf16_t* fWd1  = ws + off; off += (size_t)kKD1  * kNTD1 * 512;
    bf16_t* fWmu  = ws + off; off += (size_t)kKMU  * kNTMU * 512;
    bf16_t* fWstd = ws + off; off += (size_t)kKMU  * kNTMU * 512;

    prep_frags<<<64, 256, 0, stream>>>(Wt1,   fWt1,  74,  256, kKT1,  kNT1,  0);
    prep_frags<<<64, 256, 0, stream>>>(Wt2,   fWt2,  256, 64,  kKT2,  kNT2,  0);
    prep_frags<<<64, 256, 0, stream>>>(W_ih0, fWih0, 138, 384, kKIH0, kNTG,  1);
    prep_frags<<<64, 256, 0, stream>>>(W_hh0, fWhh0, 128, 384, kKHH,  kNTG,  1);
    prep_frags<<<64, 256, 0, stream>>>(W_ih1, fWih1, 128, 384, kKHH,  kNTG,  1);
    prep_frags<<<64, 256, 0, stream>>>(W_hh1, fWhh1, 128, 384, kKHH,  kNTG,  1);
    prep_frags<<<64, 256, 0, stream>>>(Wd1,   fWd1,  266, 256, kKD1,  kNTD1, 0);
    prep_frags<<<64, 256, 0, stream>>>(Wmu,   fWmu,  256, 64,  kKMU,  kNTMU, 0);
    prep_frags<<<64, 256, 0, stream>>>(Wstd,  fWstd, 256, 64,  kKMU,  kNTMU, 0);

    float* out_mu  = (float*)d_out;
    float* out_std = out_mu  + (size_t)kB * kT * kOB;
    float* out_ret = out_std + (size_t)kB * kT * kOB;
    float* out_fin = out_ret + (size_t)kB * kT * 320;

    dyna_rnn_kernel<<<kB / 16, 256, 0, stream>>>(
        ob_emb, ac, noise, prevh, seq,
        b_ih0, b_hh0, b_ih1, b_hh1, bt1, bt2, bd1, bmu, bstd,
        fWt1, fWt2, fWih0, fWhh0, fWih1, fWhh1, fWd1, fWmu, fWstd,
        out_mu, out_std, out_ret, out_fin);

    (void)in_sizes; (void)n_in; (void)out_size; (void)ws_size;
}
